// _RPN_42777874268461
// MI455X (gfx1250) — compile-verified, matrix operand order checked by reading, not run on hardware
//
#include <hip/hip_runtime.h>
#include <hip/hip_bf16.h>

typedef __attribute__((ext_vector_type(16))) _Float16 v16h;
typedef __attribute__((ext_vector_type(8)))  float    v8f;

#define Bn 4
#define Cc 512
#define Hh 50
#define Ww 76
#define HWp 3800
#define NPOS 15200      // Bn*HWp
#define Aa 9
#define NANCH 34200     // HWp*Aa
#define KC 4608         // Cc*9
#define NKC (KC/32)     // 144 K-tiles for conv
#define NKH (512/32)    // 16 K-tiles for heads
#define PRE_NMS 6000
#define POST_NMS 300
#define NMS_THR 0.7f

// Base anchors from _generate_anchors() (scales 8/16/32, ratios .5/1/2), computed by hand.
__constant__ float BASE_ANCH[9][4] = {
  { -84.f,  -40.f,  99.f,  55.f},
  {-176.f,  -88.f, 191.f, 103.f},
  {-360.f, -184.f, 375.f, 199.f},
  { -56.f,  -56.f,  71.f,  71.f},
  {-120.f, -120.f, 135.f, 135.f},
  {-248.f, -248.f, 263.f, 263.f},
  { -36.f,  -80.f,  51.f,  95.f},
  { -80.f, -168.f,  95.f, 183.f},
  {-168.f, -344.f, 183.f, 359.f},
};

__device__ __forceinline__ float urand(unsigned b, unsigned i, unsigned salt) {
  unsigned x = i * 1664525u + b * 1013904223u + salt * 2654435761u;
  x ^= x >> 16; x *= 0x7feb352du; x ^= x >> 15; x *= 0x846ca68bu; x ^= x >> 16;
  return (float)(x >> 8) * (1.0f / 16777216.0f);
}

__device__ __forceinline__ void atomicMaxFloat(float* addr, float val) {
  unsigned* ua = (unsigned*)addr;
  unsigned old = *ua;
  while (true) {
    float f = __uint_as_float(old);
    if (f >= val) break;
    unsigned assumed = old;
    old = atomicCAS(ua, assumed, __float_as_uint(val));
    if (old == assumed) break;
  }
}

__device__ __forceinline__ float iou_masked(float ax1, float ay1, float ax2, float ay2,
                                            float aarea, const float* g, int inside) {
  float gx1 = g[0], gy1 = g[1], gx2 = g[2], gy2 = g[3];
  float gw = gx2 - gx1 + 1.0f, gh = gy2 - gy1 + 1.0f;
  float iw = fmaxf(fminf(ax2, gx2) - fmaxf(ax1, gx1) + 1.0f, 0.0f);
  float ih = fmaxf(fminf(ay2, gy2) - fmaxf(ay1, gy1) + 1.0f, 0.0f);
  float inter = iw * ih;
  float iou = inter / (aarea + gw * gh - inter);
  if (gw == 1.0f && gh == 1.0f) iou = 0.0f;
  if (!inside) iou = -1.0f;
  return iou;
}

// ---------------- init / weight prep ----------------

__global__ void k_init(float* gt_max, float* lsum) {
  int t = threadIdx.x;
  if (t < Bn * 20) gt_max[t] = -1e30f;
  if (t < 8) lsum[t] = 0.0f;
}

// Reorder conv weights OIHW -> [co][(kh*3+kw)*512 + ci], f16.
// With K ordered plane-major, every aligned 32-wide K-tile has a single (kh,kw).
__global__ void k_wconv(const float* __restrict__ cw, _Float16* __restrict__ wf16) {
  int gid = blockIdx.x * blockDim.x + threadIdx.x;
  if (gid >= Cc * KC) return;
  int co = gid / KC, r = gid % KC;
  int q = r >> 9, ci = r & 511;          // q = kh*3+kw
  wf16[gid] = (_Float16)cw[(size_t)co * KC + ci * 9 + q];
}

__global__ void k_wprep(const float* __restrict__ cls_w, const float* __restrict__ bbox_w,
                        const float* __restrict__ cls_b, const float* __restrict__ bbox_b,
                        _Float16* __restrict__ w1, float* __restrict__ bias1) {
  int gid = blockIdx.x * blockDim.x + threadIdx.x;
  if (gid >= 64 * 512) return;
  int r = gid >> 9, k = gid & 511;
  float v = 0.0f;
  if (r < 18) v = cls_w[r * 512 + k];
  else if (r < 54) v = bbox_w[(r - 18) * 512 + k];
  w1[gid] = (_Float16)v;
  if (gid < 64) {
    float bv = 0.0f;
    if (gid < 18) bv = cls_b[gid];
    else if (gid < 54) bv = bbox_b[gid - 18];
    bias1[gid] = bv;
  }
}

// ---------------- WMMA implicit-GEMM 3x3 conv + ReLU ----------------
// M=512 (co), N=15200 (b,h,w), K=4608 plane-major. BM=64, BN=128.
// 8 waves = 4(M) x 2(N); each wave: 16x64 tile = 4 WMMAs/K-step, A frag reused 4x.
// Double-buffered LDS: one barrier per K-step.

__global__ void k_conv(const float* __restrict__ xin, const _Float16* __restrict__ wf16,
                       const float* __restrict__ bias, _Float16* __restrict__ xout) {
  __shared__ _Float16 As[2][64][40];
  __shared__ _Float16 Bs[2][128][40];
  const int tid  = threadIdx.x;
  const int wave = tid >> 5, lane = tid & 31;
  const int hv = lane >> 4, mr = lane & 15;   // half-wave, row-within-16
  const int wm = wave & 3, wn = wave >> 2;
  const int m0 = blockIdx.y * 64;
  const int n0 = blockIdx.x * 128;

  // hoisted per-thread A staging coords
  const int aidx = tid * 8;
  const int amrow = aidx >> 5, akk = aidx & 31;
  // hoisted per-thread B staging coords (n-derived values are kt-invariant)
  const int nrowB = tid >> 1, kk0B = (tid & 1) * 16;
  const int nB = n0 + nrowB;
  const int bbB = nB / HWp, ppB = nB % HWp;
  const int hB = ppB / Ww, wB = ppB % Ww;
  const bool nok = (nB < NPOS);

  v8f acc[4] = {};

  auto stageA = [&](int kt, int buf) {
    const _Float16* src = wf16 + (size_t)(m0 + amrow) * KC + kt * 32 + akk;
    if (kt + 2 < NKC) __builtin_prefetch((const void*)(src + 64), 0, 1);
#pragma unroll
    for (int e = 0; e < 8; ++e) As[buf][amrow][akk + e] = src[e];
  };
  auto stageB = [&](int kt, int buf) {
    int q = kt >> 4;                 // (kh,kw) plane; constant within the tile
    int ci0 = (kt & 15) << 5;
    int kh = q / 3, kw = q - kh * 3;
    int hi2 = hB + kh - 1, wi2 = wB + kw - 1;
    bool ok = nok && hi2 >= 0 && hi2 < Hh && wi2 >= 0 && wi2 < Ww;
    const float* src = xin + (((size_t)bbB * Cc + (ci0 + kk0B)) * Hh + hi2) * Ww + wi2;
#pragma unroll
    for (int e = 0; e < 16; ++e) {
      float v = ok ? src[(size_t)e * (Hh * Ww)] : 0.0f;
      Bs[buf][nrowB][kk0B + e] = (_Float16)v;
    }
  };
  auto compute = [&](int buf) {
    v16h af;
#pragma unroll
    for (int e = 0; e < 8; ++e) {
      af[e]     = As[buf][wm * 16 + mr][hv * 8 + e];
      af[e + 8] = As[buf][wm * 16 + mr][16 + hv * 8 + e];
    }
#pragma unroll
    for (int sub = 0; sub < 4; ++sub) {
      v16h bf;
#pragma unroll
      for (int e = 0; e < 8; ++e) {
        bf[e]     = Bs[buf][wn * 64 + sub * 16 + mr][hv * 8 + e];
        bf[e + 8] = Bs[buf][wn * 64 + sub * 16 + mr][16 + hv * 8 + e];
      }
      acc[sub] = __builtin_amdgcn_wmma_f32_16x16x32_f16(false, af, false, bf,
                                                        (short)0, acc[sub], false, false);
    }
  };

  stageA(0, 0); stageB(0, 0);
  __syncthreads();
  for (int kt = 0; kt < NKC; ++kt) {
    int cur = kt & 1;
    if (kt + 1 < NKC) { stageA(kt + 1, cur ^ 1); stageB(kt + 1, cur ^ 1); }
    compute(cur);
    __syncthreads();
  }

#pragma unroll
  for (int sub = 0; sub < 4; ++sub) {
    int n = n0 + wn * 64 + sub * 16 + mr;
#pragma unroll
    for (int r = 0; r < 8; ++r) {
      int m = m0 + wm * 16 + r + hv * 8;
      float v = acc[sub][r] + bias[m];
      v = fmaxf(v, 0.0f);
      if (n < NPOS) xout[(size_t)n * Cc + m] = (_Float16)v;
    }
  }
}

// ---------------- WMMA 1x1 heads (cls 18 + bbox 36 fused, padded to 64) ----------------

__global__ void k_head(const _Float16* __restrict__ xf16, const _Float16* __restrict__ w1,
                       const float* __restrict__ bias1, float* __restrict__ out1) {
  __shared__ _Float16 As[2][64][40];
  __shared__ _Float16 Bs[2][128][40];
  const int tid  = threadIdx.x;
  const int wave = tid >> 5, lane = tid & 31;
  const int hv = lane >> 4, mr = lane & 15;
  const int wm = wave & 3, wn = wave >> 2;
  const int n0 = blockIdx.x * 128;

  const int aidx = tid * 8;
  const int amrow = aidx >> 5, akk = aidx & 31;
  const int nrowB = tid >> 1, kk0B = (tid & 1) * 16;
  const int nB = n0 + nrowB;
  const bool nok = (nB < NPOS);

  v8f acc[4] = {};

  auto stageA = [&](int kt, int buf) {
    const _Float16* src = w1 + (size_t)amrow * 512 + kt * 32 + akk;
#pragma unroll
    for (int e = 0; e < 8; ++e) As[buf][amrow][akk + e] = src[e];
  };
  auto stageB = [&](int kt, int buf) {
    const _Float16* src = xf16 + (size_t)nB * 512 + kt * 32 + kk0B;
#pragma unroll
    for (int e = 0; e < 16; ++e)
      Bs[buf][nrowB][kk0B + e] = nok ? src[e] : (_Float16)0.0f;
  };
  auto compute = [&](int buf) {
    v16h af;
#pragma unroll
    for (int e = 0; e < 8; ++e) {
      af[e]     = As[buf][wm * 16 + mr][hv * 8 + e];
      af[e + 8] = As[buf][wm * 16 + mr][16 + hv * 8 + e];
    }
#pragma unroll
    for (int sub = 0; sub < 4; ++sub) {
      v16h bf;
#pragma unroll
      for (int e = 0; e < 8; ++e) {
        bf[e]     = Bs[buf][wn * 64 + sub * 16 + mr][hv * 8 + e];
        bf[e + 8] = Bs[buf][wn * 64 + sub * 16 + mr][16 + hv * 8 + e];
      }
      acc[sub] = __builtin_amdgcn_wmma_f32_16x16x32_f16(false, af, false, bf,
                                                        (short)0, acc[sub], false, false);
    }
  };

  stageA(0, 0); stageB(0, 0);
  __syncthreads();
  for (int kt = 0; kt < NKH; ++kt) {
    int cur = kt & 1;
    if (kt + 1 < NKH) { stageA(kt + 1, cur ^ 1); stageB(kt + 1, cur ^ 1); }
    compute(cur);
    __syncthreads();
  }

#pragma unroll
  for (int sub = 0; sub < 4; ++sub) {
    int n = n0 + wn * 64 + sub * 16 + mr;
#pragma unroll
    for (int r = 0; r < 8; ++r) {
      int m = wm * 16 + r + hv * 8;
      if (n < NPOS) out1[(size_t)n * 64 + m] = acc[sub][r] + bias1[m];
    }
  }
}

// ---------------- proposal: scores + decoded clipped boxes ----------------

__global__ void k_score(const float* __restrict__ out1, const float* __restrict__ iminfo,
                        float* __restrict__ scores, float* __restrict__ boxes) {
  int gid = blockIdx.x * blockDim.x + threadIdx.x;
  if (gid >= Bn * NANCH) return;
  int b = gid / NANCH, i = gid % NANCH;
  int p = i / Aa, a = i % Aa;
  int h = p / Ww, w = p % Ww;
  float imh = iminfo[b * 3 + 0], imw = iminfo[b * 3 + 1];
  const float* row = out1 + (size_t)(b * HWp + p) * 64;
  float s0 = row[a], s1 = row[9 + a];
  scores[gid] = 1.0f / (1.0f + expf(s0 - s1));   // softmax fg prob
  float d0 = row[18 + a * 4 + 0], d1 = row[18 + a * 4 + 1];
  float d2 = row[18 + a * 4 + 2], d3 = row[18 + a * 4 + 3];
  float sx = (float)w * 16.0f, sy = (float)h * 16.0f;
  float ax1 = BASE_ANCH[a][0] + sx, ay1 = BASE_ANCH[a][1] + sy;
  float ax2 = BASE_ANCH[a][2] + sx, ay2 = BASE_ANCH[a][3] + sy;
  float aw = ax2 - ax1 + 1.0f, ah = ay2 - ay1 + 1.0f;
  float cx = ax1 + 0.5f * aw, cy = ay1 + 0.5f * ah;
  float pcx = d0 * aw + cx, pcy = d1 * ah + cy;
  float pw = expf(d2) * aw, ph = expf(d3) * ah;
  float x1 = fminf(fmaxf(pcx - 0.5f * pw, 0.0f), imw - 1.0f);
  float y1 = fminf(fmaxf(pcy - 0.5f * ph, 0.0f), imh - 1.0f);
  float x2 = fminf(fmaxf(pcx + 0.5f * pw, 0.0f), imw - 1.0f);
  float y2 = fminf(fmaxf(pcy + 0.5f * ph, 0.0f), imh - 1.0f);
  float* o = boxes + (size_t)gid * 4;
  o[0] = x1; o[1] = y1; o[2] = x2; o[3] = y2;
}

// 6000th-largest score via binary search (scores in (0,1))
__global__ void k_thresh(const float* __restrict__ scores, float* __restrict__ thr) {
  int b = blockIdx.x, tid = threadIdx.x;
  const float* S = scores + (size_t)b * NANCH;
  __shared__ int sred[256];
  float lo = 0.0f, hi = 1.0f;
  for (int it = 0; it < 30; ++it) {
    float mid = 0.5f * (lo + hi);
    int c = 0;
    for (int i = tid; i < NANCH; i += 256) if (S[i] > mid) c++;
    sred[tid] = c; __syncthreads();
    for (int o = 128; o; o >>= 1) { if (tid < o) sred[tid] += sred[tid + o]; __syncthreads(); }
    int tot = sred[0]; __syncthreads();
    if (tot >= PRE_NMS) lo = mid; else hi = mid;
  }
  if (tid == 0) thr[b] = lo;
}

// iterative NMS, one block per batch; writes rois into d_out
__global__ void k_nms(const float* __restrict__ scores, const float* __restrict__ boxes,
                      const float* __restrict__ thr, int* __restrict__ valid,
                      float* __restrict__ out) {
  int b = blockIdx.x, tid = threadIdx.x;
  const float* S = scores + (size_t)b * NANCH;
  const float* BX = boxes + (size_t)b * NANCH * 4;
  int* V = valid + (size_t)b * NANCH;
  float t = thr[b];
  for (int i = tid; i < NANCH; i += 256) V[i] = (S[i] > t) ? 1 : 0;
  __syncthreads();
  __shared__ float sv[256];
  __shared__ int si[256];
  for (int k = 0; k < POST_NMS; ++k) {
    float best = -1e30f; int bi = -1;
    for (int i = tid; i < NANCH; i += 256)
      if (V[i] && S[i] > best) { best = S[i]; bi = i; }
    sv[tid] = best; si[tid] = bi;
    __syncthreads();
    for (int o = 128; o; o >>= 1) {
      if (tid < o) {
        float ov = sv[tid + o]; int oi = si[tid + o];
        if (oi >= 0 && (si[tid] < 0 || ov > sv[tid] || (ov == sv[tid] && oi < si[tid]))) {
          sv[tid] = ov; si[tid] = oi;
        }
      }
      __syncthreads();
    }
    int sel = si[0];
    float bx1 = 0, by1 = 0, bx2 = 0, by2 = 0;
    if (sel >= 0) { bx1 = BX[sel * 4]; by1 = BX[sel * 4 + 1]; bx2 = BX[sel * 4 + 2]; by2 = BX[sel * 4 + 3]; }
    if (tid == 0) {
      float* o = out + ((size_t)b * POST_NMS + k) * 5;
      o[0] = (float)b; o[1] = bx1; o[2] = by1; o[3] = bx2; o[4] = by2;
    }
    if (sel >= 0) {
      float as_ = (bx2 - bx1 + 1.0f) * (by2 - by1 + 1.0f);
      for (int i = tid; i < NANCH; i += 256) {
        if (!V[i]) continue;
        float x1 = BX[i * 4], y1 = BX[i * 4 + 1], x2 = BX[i * 4 + 2], y2 = BX[i * 4 + 3];
        float iw = fmaxf(fminf(bx2, x2) - fmaxf(bx1, x1) + 1.0f, 0.0f);
        float ih = fmaxf(fminf(by2, y2) - fmaxf(by1, y1) + 1.0f, 0.0f);
        float inter = iw * ih;
        float ai = (x2 - x1 + 1.0f) * (y2 - y1 + 1.0f);
        if (inter / (as_ + ai - inter) > NMS_THR) V[i] = 0;
      }
    }
    __syncthreads();
  }
}

// ---------------- anchor targets ----------------

__global__ void k_at1(const float* __restrict__ gt, const float* __restrict__ iminfo,
                      float* __restrict__ max_ov, int* __restrict__ argg,
                      int* __restrict__ insd, float* __restrict__ gt_max) {
  int gid = blockIdx.x * blockDim.x + threadIdx.x;
  if (gid >= Bn * NANCH) return;
  int b = gid / NANCH, i = gid % NANCH;
  int p = i / Aa, a = i % Aa;
  int h = p / Ww, w = p % Ww;
  float imh = iminfo[b * 3 + 0], imw = iminfo[b * 3 + 1];
  float sx = (float)w * 16.0f, sy = (float)h * 16.0f;
  float ax1 = BASE_ANCH[a][0] + sx, ay1 = BASE_ANCH[a][1] + sy;
  float ax2 = BASE_ANCH[a][2] + sx, ay2 = BASE_ANCH[a][3] + sy;
  int inside = (ax1 >= 0.0f && ay1 >= 0.0f && ax2 < imw && ay2 < imh);
  float aarea = (ax2 - ax1 + 1.0f) * (ay2 - ay1 + 1.0f);
  float best = -1e30f; int barg = 0;
  for (int j = 0; j < 20; ++j) {
    float v = iou_masked(ax1, ay1, ax2, ay2, aarea, gt + (size_t)(b * 20 + j) * 5, inside);
    atomicMaxFloat(&gt_max[b * 20 + j], v);
    if (v > best) { best = v; barg = j; }
  }
  max_ov[gid] = best; argg[gid] = barg; insd[gid] = inside;
}

__global__ void k_at2(const float* __restrict__ gt, const float* __restrict__ iminfo,
                      const float* __restrict__ max_ov, const float* __restrict__ gt_max,
                      int* __restrict__ labelp) {
  int gid = blockIdx.x * blockDim.x + threadIdx.x;
  if (gid >= Bn * NANCH) return;
  int b = gid / NANCH, i = gid % NANCH;
  int p = i / Aa, a = i % Aa;
  int h = p / Ww, w = p % Ww;
  float imh = iminfo[b * 3 + 0], imw = iminfo[b * 3 + 1];
  float sx = (float)w * 16.0f, sy = (float)h * 16.0f;
  float ax1 = BASE_ANCH[a][0] + sx, ay1 = BASE_ANCH[a][1] + sy;
  float ax2 = BASE_ANCH[a][2] + sx, ay2 = BASE_ANCH[a][3] + sy;
  int inside = (ax1 >= 0.0f && ay1 >= 0.0f && ax2 < imw && ay2 < imh);
  float aarea = (ax2 - ax1 + 1.0f) * (ay2 - ay1 + 1.0f);
  float mo = max_ov[gid];
  int label = -1;
  if (mo < 0.3f) label = 0;
  bool keep = false;
  for (int j = 0; j < 20; ++j) {
    float v = iou_masked(ax1, ay1, ax2, ay2, aarea, gt + (size_t)(b * 20 + j) * 5, inside);
    float gm = gt_max[b * 20 + j];
    gm = (gm <= 0.0f) ? 1e-5f : gm;
    if (v == gm) keep = true;
  }
  if (keep) label = 1;
  if (mo >= 0.7f) label = 1;
  if (!inside) label = -1;
  labelp[gid] = label;
}

// subsample thresholds per batch
__global__ void k_sub(const int* __restrict__ labelp, float* __restrict__ subs) {
  int b = blockIdx.x, tid = threadIdx.x;
  const int* L = labelp + (size_t)b * NANCH;
  __shared__ int sred[256];
  auto count = [&](int want, float t, unsigned salt) -> int {
    int c = 0;
    for (int i = tid; i < NANCH; i += 256)
      if (L[i] == want && urand(b, i, salt) < t) c++;
    sred[tid] = c; __syncthreads();
    for (int o = 128; o; o >>= 1) { if (tid < o) sred[tid] += sred[tid + o]; __syncthreads(); }
    int tot = sred[0]; __syncthreads();
    return tot;
  };
  int cnt_fg = count(1, 2.0f, 17u);
  float t_fg = 2.0f;
  if (cnt_fg > 128) {
    float lo = 0.0f, hi = 1.0f;
    for (int it = 0; it < 24; ++it) {
      float mid = 0.5f * (lo + hi);
      int c = count(1, mid, 17u);
      if (c < 128) lo = mid; else hi = mid;
    }
    t_fg = hi;
  }
  int kept_fg = (t_fg >= 2.0f) ? cnt_fg : count(1, t_fg, 17u);
  int num_bg = 256 - kept_fg;
  int cnt_bg = count(0, 2.0f, 91u);
  float t_bg = 2.0f;
  if (cnt_bg > num_bg && num_bg > 0) {
    float lo = 0.0f, hi = 1.0f;
    for (int it = 0; it < 24; ++it) {
      float mid = 0.5f * (lo + hi);
      int c = count(0, mid, 91u);
      if (c < num_bg) lo = mid; else hi = mid;
    }
    t_bg = hi;
  }
  int kept_bg = (t_bg >= 2.0f) ? cnt_bg : count(0, t_bg, 91u);
  if (tid == 0) {
    float nex = (float)(kept_fg + kept_bg);
    if (nex < 1.0f) nex = 1.0f;
    subs[b * 4 + 0] = t_fg; subs[b * 4 + 1] = t_bg; subs[b * 4 + 2] = nex;
  }
}

__global__ void k_loss(const float* __restrict__ out1, const int* __restrict__ labelp,
                       const int* __restrict__ argg, const float* __restrict__ gt,
                       const float* __restrict__ subs, float* __restrict__ lsum) {
  int gid = blockIdx.x * blockDim.x + threadIdx.x;
  if (gid >= Bn * NANCH) return;
  int b = gid / NANCH, i = gid % NANCH;
  int lp = labelp[gid];
  float tfg = subs[b * 4 + 0], tbg = subs[b * 4 + 1], nex = subs[b * 4 + 2];
  int label = -1;
  if (lp == 1 && urand(b, i, 17u) < tfg) label = 1;
  else if (lp == 0 && urand(b, i, 91u) < tbg) label = 0;
  if (label < 0) return;
  int p = i / Aa, a = i % Aa;
  const float* row = out1 + (size_t)(b * HWp + p) * 64;
  float s0 = row[a], s1 = row[9 + a];
  float mx = fmaxf(s0, s1);
  float lse = mx + logf(expf(s0 - mx) + expf(s1 - mx));
  float nll = lse - (label == 1 ? s1 : s0);
  atomicAdd(&lsum[0], nll);
  atomicAdd(&lsum[1], 1.0f);
  if (label == 1) {
    int h = p / Ww, w = p % Ww;
    float sx = (float)w * 16.0f, sy = (float)h * 16.0f;
    float ax1 = BASE_ANCH[a][0] + sx, ay1 = BASE_ANCH[a][1] + sy;
    float ax2 = BASE_ANCH[a][2] + sx, ay2 = BASE_ANCH[a][3] + sy;
    const float* g = gt + (size_t)(b * 20 + argg[gid]) * 5;
    float ew = ax2 - ax1 + 1.0f, eh = ay2 - ay1 + 1.0f;
    float ecx = ax1 + 0.5f * ew, ecy = ay1 + 0.5f * eh;
    float gw = g[2] - g[0] + 1.0f, gh = g[3] - g[1] + 1.0f;
    float gcx = g[0] + 0.5f * gw, gcy = g[1] + 0.5f * gh;
    float tg[4] = {(gcx - ecx) / ew, (gcy - ecy) / eh, logf(gw / ew), logf(gh / eh)};
    float s = 0.0f;
#pragma unroll
    for (int k = 0; k < 4; ++k) {
      float d = row[18 + a * 4 + k] - tg[k];
      float ad = fabsf(d);
      s += (ad < 1.0f / 9.0f) ? 0.5f * 9.0f * d * d : ad - 0.5f / 9.0f;
    }
    atomicAdd(&lsum[2 + b], s / nex);
  }
}

__global__ void k_fin(const float* __restrict__ lsum, float* __restrict__ out) {
  out[Bn * POST_NMS * 5 + 0] = lsum[0] / fmaxf(lsum[1], 1.0f);
  out[Bn * POST_NMS * 5 + 1] = 0.25f * (lsum[2] + lsum[3] + lsum[4] + lsum[5]);
}

// ---------------- launch ----------------

extern "C" void kernel_launch(void* const* d_in, const int* in_sizes, int n_in,
                              void* d_out, int out_size, void* d_ws, size_t ws_size,
                              hipStream_t stream) {
  const float* base_feat = (const float*)d_in[0];
  const float* im_info   = (const float*)d_in[1];
  const float* gt        = (const float*)d_in[2];
  const float* conv_w    = (const float*)d_in[4];
  const float* conv_b    = (const float*)d_in[5];
  const float* cls_w     = (const float*)d_in[6];
  const float* cls_b     = (const float*)d_in[7];
  const float* bbox_w    = (const float*)d_in[8];
  const float* bbox_b    = (const float*)d_in[9];
  float* out = (float*)d_out;
  (void)in_sizes; (void)n_in; (void)out_size; (void)ws_size;

  char* base = (char*)d_ws;
  size_t off = 0;
  auto alloc = [&](size_t bytes) -> void* {
    void* p = base + off;
    off = (off + bytes + 255) & ~(size_t)255;
    return p;
  };
  _Float16* wf16  = (_Float16*)alloc((size_t)Cc * KC * 2);
  _Float16* w1f16 = (_Float16*)alloc((size_t)64 * 512 * 2);
  float* bias1    = (float*)alloc(64 * 4);
  _Float16* xf16  = (_Float16*)alloc((size_t)NPOS * Cc * 2);
  float* out1     = (float*)alloc((size_t)NPOS * 64 * 4);
  float* scores   = (float*)alloc((size_t)Bn * NANCH * 4);
  float* boxes    = (float*)alloc((size_t)Bn * NANCH * 16);
  int*   valid    = (int*)alloc((size_t)Bn * NANCH * 4);
  float* thr      = (float*)alloc(Bn * 4);
  float* max_ov   = (float*)alloc((size_t)Bn * NANCH * 4);
  int*   argg     = (int*)alloc((size_t)Bn * NANCH * 4);
  int*   insd     = (int*)alloc((size_t)Bn * NANCH * 4);
  int*   labelp   = (int*)alloc((size_t)Bn * NANCH * 4);
  float* gt_max   = (float*)alloc(Bn * 20 * 4);
  float* subs     = (float*)alloc(Bn * 4 * 4);
  float* lsum     = (float*)alloc(8 * 4);

  const int NA = Bn * NANCH;
  const int NBLK = (NPOS + 127) / 128;   // 119 N-tiles of 128
  k_init<<<1, 256, 0, stream>>>(gt_max, lsum);
  k_wconv<<<(Cc * KC + 255) / 256, 256, 0, stream>>>(conv_w, wf16);
  k_wprep<<<(64 * 512 + 255) / 256, 256, 0, stream>>>(cls_w, bbox_w, cls_b, bbox_b, w1f16, bias1);
  k_conv<<<dim3(NBLK, Cc / 64), 256, 0, stream>>>(base_feat, wf16, conv_b, xf16);
  k_head<<<dim3(NBLK, 1), 256, 0, stream>>>(xf16, w1f16, bias1, out1);
  k_score<<<(NA + 255) / 256, 256, 0, stream>>>(out1, im_info, scores, boxes);
  k_thresh<<<Bn, 256, 0, stream>>>(scores, thr);
  k_nms<<<Bn, 256, 0, stream>>>(scores, boxes, thr, valid, out);
  k_at1<<<(NA + 255) / 256, 256, 0, stream>>>(gt, im_info, max_ov, argg, insd, gt_max);
  k_at2<<<(NA + 255) / 256, 256, 0, stream>>>(gt, im_info, max_ov, gt_max, labelp);
  k_sub<<<Bn, 256, 0, stream>>>(labelp, subs);
  k_loss<<<(NA + 255) / 256, 256, 0, stream>>>(out1, labelp, argg, gt, subs, lsum);
  k_fin<<<1, 1, 0, stream>>>(lsum, out);
  (void)insd;
}
